// MoE_9912784519966
// MI455X (gfx1250) — matrix operand order repaired and checked by reading, not compiled, not run on hardware
//
#include <hip/hip_runtime.h>
#include <hip/hip_bf16.h>

// MI455X (gfx1250, wave32) MoE pipeline.
// Convs: implicit GEMM on v_wmma_f32_16x16x32_f16 (f16 in, f32 acc).
// K-loop fully unrolled so the im2col decode folds into immediate LDS
// offsets (ds_load_u16 + base) instead of per-element VALU address math.
// Pools / router / expert GEMVs: bandwidth-bound VALU kernels.

typedef _Float16 v16h __attribute__((ext_vector_type(16)));
typedef float    v8f  __attribute__((ext_vector_type(8)));

// ---------------------------------------------------------------------------
// 3x3 SAME conv + bias + ReLU as implicit GEMM via WMMA f16.
//   M = COUT (16-wide tiles), N = pixels (16-wide tiles), K = CIN*9 (padded
//   to multiple of 32). 256 threads = 8 waves; waves tile (cout, pixel).
// ---------------------------------------------------------------------------
template<int CIN, int COUT, int H, int W>
__global__ void __launch_bounds__(256)
conv3x3_wmma(const _Float16* __restrict__ in,
             const _Float16* __restrict__ wpk,   // packed [COUT][KPAD] f16
             const float*    __restrict__ bias,
             _Float16*       __restrict__ out)
{
    constexpr int KREAL = CIN * 9;
    constexpr int KPAD  = ((KREAL + 31) / 32) * 32;
    constexpr int NCT   = COUT / 16;   // cout tiles per block (2 or 4)
    constexpr int NPT   = 8 / NCT;     // pixel tiles per block
    constexpr int PIX   = NPT * 16;    // pixels per block (one row span)
    constexpr int LW    = PIX + 2;     // halo width

    __shared__ _Float16 sIn[CIN * 3 * LW];

    const int wtiles = W / PIX;
    const int bt = blockIdx.x;
    const int wt = bt % wtiles;
    const int h  = (bt / wtiles) % H;
    const int b  = bt / (wtiles * H);
    const int w0 = wt * PIX;
    const int tid = threadIdx.x;

    // Stage input halo patch (rows h-1..h+1, cols w0-1..w0+PIX), zero-padded.
    for (int idx = tid; idx < CIN * 3 * LW; idx += 256) {
        int ci  = idx / (3 * LW);
        int rem = idx - ci * 3 * LW;
        int r   = rem / LW;
        int cc  = rem - r * LW;
        int ih  = h + r - 1;
        int iw  = w0 + cc - 1;
        _Float16 v = (_Float16)0.0f;
        if (ih >= 0 && ih < H && iw >= 0 && iw < W)
            v = in[(((size_t)b * CIN + ci) * H + ih) * W + iw];
        sIn[idx] = v;
    }
    __syncthreads();

    const int wave = tid >> 5;
    const int lane = tid & 31;
    const int hi   = lane >> 4;            // K-half select (wave32 WMMA layout)
    const int l15  = lane & 15;
    const int cout_base = (wave % NCT) * 16;
    const int wl        = (wave / NCT) * 16 + l15;   // local pixel (B column)

    v8f acc = {};
    const _Float16* __restrict__ arow = wpk + (cout_base + l15) * KPAD + hi * 8;
    const _Float16* __restrict__ sB   = sIn + wl;    // per-lane LDS base

    // Fully unrolled K loop: every k below is a compile-time constant, so the
    // im2col decode becomes immediate ds_load_u16 offsets and the A loads
    // become clause-able global_load_b128 with immediate offsets.
    #pragma unroll
    for (int k0 = 0; k0 < KPAD; k0 += 32) {
        // A fragment: per 16-bit A 16x32 layout each lane holds two
        // contiguous 8-half runs -> two aligned 16B global loads.
        union { uint4 q[2]; v16h v; } au;
        au.q[0] = *reinterpret_cast<const uint4*>(arow + k0);
        au.q[1] = *reinterpret_cast<const uint4*>(arow + k0 + 16);

        // B fragment: im2col decode from LDS (lane holds column n, K=k0+8*hi+e for e<8, k0+8+8*hi+e for e>=8).
        union { _Float16 e[16]; v16h v; } bu;
        #pragma unroll
        for (int ee = 0; ee < 16; ++ee) {
            const int kA = k0 + (ee < 8 ? ee : 8 + ee);         // variant taken when hi == 0
            const int kB = k0 + 8 + (ee < 8 ? ee : 8 + ee);    // variant taken when hi == 1
            _Float16 vA = (_Float16)0.0f, vB = (_Float16)0.0f;
            if (kA < KREAL) {
                constexpr int dummy = 0; (void)dummy;
                const int ci = kA / 9, r9 = kA % 9, r = r9 / 3, c = r9 % 3;
                vA = sB[(ci * 3 + r) * LW + c];
            }
            if (kB < KREAL) {
                const int ci = kB / 9, r9 = kB % 9, r = r9 / 3, c = r9 % 3;
                vB = sB[(ci * 3 + r) * LW + c];
            }
            bu.e[ee] = hi ? vB : vA;
        }

        acc = __builtin_amdgcn_wmma_f32_16x16x32_f16(
                false, au.v, false, bu.v, (short)0, acc, false, false);
    }

    // C/D layout: VGPR r holds M = r + 8*hi, N = lane&15.
    // One base address; per-r stores use constant H*W stride (fits the
    // 24-bit signed instruction offset for every layer here).
    const int wpix  = w0 + wl;
    const int cout0 = cout_base + hi * 8;
    _Float16* __restrict__ op = out + (((size_t)b * COUT + cout0) * H + h) * W + wpix;
    const float* __restrict__ bp = bias + cout0;
    #pragma unroll
    for (int r = 0; r < 8; ++r) {
        float v = acc[r] + bp[r];
        v = v > 0.0f ? v : 0.0f;
        op[(size_t)r * H * W] = (_Float16)v;
    }
}

// ---------------------------------------------------------------------------
// Helpers
// ---------------------------------------------------------------------------
__global__ void f32_to_f16(const float* __restrict__ in, _Float16* __restrict__ out, int n)
{
    int i = blockIdx.x * 256 + threadIdx.x;
    if (i < n) out[i] = (_Float16)in[i];
}

__global__ void pack_weights(const float* __restrict__ w, _Float16* __restrict__ out,
                             int COUT, int KREAL, int KPAD)
{
    int i = blockIdx.x * 256 + threadIdx.x;
    if (i >= COUT * KPAD) return;
    int co = i / KPAD, k = i - co * KPAD;
    out[i] = (_Float16)((k < KREAL) ? w[co * KREAL + k] : 0.0f);
}

__global__ void avgpool2_f16(const _Float16* __restrict__ in, _Float16* __restrict__ out,
                             int C, int H, int W, int total)
{
    int idx = blockIdx.x * 256 + threadIdx.x;
    if (idx >= total) return;
    int Wo = W >> 1, Ho = H >> 1;
    int wo = idx % Wo;
    int t  = idx / Wo;
    int ho = t % Ho; t /= Ho;
    int c  = t % C;
    int b  = t / C;
    size_t base = (((size_t)b * C + c) * H + 2 * ho) * W + 2 * wo;
    float s = (float)in[base] + (float)in[base + 1]
            + (float)in[base + W] + (float)in[base + W + 1];
    out[idx] = (_Float16)(0.25f * s);
}

// Global mean over 32x32 per (b, c): one block per channel.
__global__ void __launch_bounds__(256)
gmean_f16(const _Float16* __restrict__ in, float* __restrict__ out)
{
    __shared__ float sred[256];
    int bc  = blockIdx.x;       // b*64 + c
    int tid = threadIdx.x;
    size_t base = (size_t)bc * 1024;
    float s = 0.0f;
    #pragma unroll
    for (int i = 0; i < 4; ++i) s += (float)in[base + tid + i * 256];
    sred[tid] = s;
    __syncthreads();
    for (int off = 128; off > 0; off >>= 1) {
        if (tid < off) sred[tid] += sred[tid + off];
        __syncthreads();
    }
    if (tid == 0) out[bc] = sred[0] * (1.0f / 1024.0f);
}

// z = pooled @ proj_w + proj_b   ([128,64] x [64,512])
__global__ void proj_kernel(const float* __restrict__ pooled, const float* __restrict__ pw,
                            const float* __restrict__ pb, float* __restrict__ z)
{
    int idx = blockIdx.x * 256 + threadIdx.x;   // 128*512
    int d = idx & 511;
    int b = idx >> 9;
    float acc = pb[d];
    #pragma unroll 8
    for (int c = 0; c < 64; ++c) acc += pooled[b * 64 + c] * pw[c * 512 + d];
    z[idx] = acc;
}

// Router: one block, one thread per token. Produces top1, aux_loss, counts,
// importance (tail of d_out).
__global__ void __launch_bounds__(128)
router_kernel(const float* __restrict__ z, const float* __restrict__ rw,
              const float* __restrict__ rb, int* __restrict__ top1,
              float* __restrict__ out_tail)
{
    __shared__ float simp[16];
    __shared__ float scnt[16];
    int t = threadIdx.x;
    if (t < 16) { simp[t] = 0.0f; scnt[t] = 0.0f; }
    __syncthreads();

    float lg[16];
    #pragma unroll
    for (int e = 0; e < 16; ++e) lg[e] = rb[e];
    for (int d = 0; d < 512; ++d) {
        float zv = z[t * 512 + d];
        #pragma unroll
        for (int e = 0; e < 16; ++e) lg[e] += zv * rw[d * 16 + e];
    }
    float m = lg[0]; int am = 0;
    #pragma unroll
    for (int e = 1; e < 16; ++e) if (lg[e] > m) { m = lg[e]; am = e; }
    float p[16], s = 0.0f;
    #pragma unroll
    for (int e = 0; e < 16; ++e) { p[e] = __expf(lg[e] - m); s += p[e]; }
    float inv = 1.0f / s;
    #pragma unroll
    for (int e = 0; e < 16; ++e) atomicAdd(&simp[e], p[e] * inv * (1.0f / 128.0f));
    atomicAdd(&scnt[am], 1.0f / 128.0f);
    top1[t] = am;
    __syncthreads();

    if (t == 0) {
        float aux = 0.0f;
        for (int e = 0; e < 16; ++e) aux += scnt[e] * simp[e];
        out_tail[0] = 16.0f * aux;              // aux_loss
    }
    if (t < 16) {
        out_tail[1 + t]  = scnt[t];             // counts
        out_tail[17 + t] = simp[t];             // importance
    }
}

// h1 = relu(z @ w1[e] + b1[e]) : one block per token, z in LDS.
__global__ void __launch_bounds__(256)
expert_fc1(const float* __restrict__ z, const float* __restrict__ w1,
           const float* __restrict__ b1, const int* __restrict__ top1,
           float* __restrict__ h1)
{
    __shared__ float sz[512];
    int b = blockIdx.x, tid = threadIdx.x;
    for (int d = tid; d < 512; d += 256) sz[d] = z[b * 512 + d];
    __syncthreads();
    int e = top1[b];
    const float* W = w1 + (size_t)e * 512 * 512;
    float a0 = b1[e * 512 + tid];
    float a1 = b1[e * 512 + tid + 256];
    for (int d = 0; d < 512; ++d) {
        float zv = sz[d];
        const float* row = W + (size_t)d * 512;
        a0 += zv * row[tid];
        a1 += zv * row[tid + 256];
    }
    h1[b * 512 + tid]       = fmaxf(a0, 0.0f);
    h1[b * 512 + tid + 256] = fmaxf(a1, 0.0f);
}

// logits = h1 @ w2[e] + b2[e] : one block per token.
__global__ void __launch_bounds__(256)
expert_fc2(const float* __restrict__ h1, const float* __restrict__ w2,
           const float* __restrict__ b2, const int* __restrict__ top1,
           float* __restrict__ logits)
{
    __shared__ float sh[512];
    int b = blockIdx.x, tid = threadIdx.x;
    for (int d = tid; d < 512; d += 256) sh[d] = h1[b * 512 + d];
    __syncthreads();
    int e = top1[b];
    const float* W = w2 + (size_t)e * 512 * 1000;
    float acc[4];
    #pragma unroll
    for (int i = 0; i < 4; ++i) {
        int n = tid + i * 256;
        acc[i] = (n < 1000) ? b2[e * 1000 + n] : 0.0f;
    }
    for (int j = 0; j < 512; ++j) {
        float hv = sh[j];
        const float* row = W + (size_t)j * 1000;
        #pragma unroll
        for (int i = 0; i < 4; ++i) {
            int n = tid + i * 256;
            if (n < 1000) acc[i] += hv * row[n];
        }
    }
    #pragma unroll
    for (int i = 0; i < 4; ++i) {
        int n = tid + i * 256;
        if (n < 1000) logits[(size_t)b * 1000 + n] = acc[i];
    }
}

// ---------------------------------------------------------------------------
extern "C" void kernel_launch(void* const* d_in, const int* in_sizes, int n_in,
                              void* d_out, int out_size, void* d_ws, size_t ws_size,
                              hipStream_t stream)
{
    (void)in_sizes; (void)n_in; (void)out_size; (void)ws_size;

    const float* x   = (const float*)d_in[0];
    const float* c1w = (const float*)d_in[1];  const float* c1b = (const float*)d_in[2];
    const float* c2w = (const float*)d_in[3];  const float* c2b = (const float*)d_in[4];
    const float* c3w = (const float*)d_in[5];  const float* c3b = (const float*)d_in[6];
    const float* c4w = (const float*)d_in[7];  const float* c4b = (const float*)d_in[8];
    const float* pw  = (const float*)d_in[9];  const float* pb  = (const float*)d_in[10];
    const float* rw  = (const float*)d_in[11]; const float* rb  = (const float*)d_in[12];
    const float* w1  = (const float*)d_in[13]; const float* b1  = (const float*)d_in[14];
    const float* w2  = (const float*)d_in[15]; const float* b2  = (const float*)d_in[16];
    float* out = (float*)d_out;

    // Workspace layout (rotating f16 activation buffers + small tensors).
    char* ws = (char*)d_ws;
    constexpr size_t SZ_BIG = 134217728;                     // 128 MB
    _Float16* bufA = (_Float16*)(ws);                        // 128 MB
    _Float16* bufB = (_Float16*)(ws + SZ_BIG);               // 128 MB
    _Float16* bufC = (_Float16*)(ws + 2 * SZ_BIG);           // 32 MB
    char* sm = ws + 2 * SZ_BIG + 33554432;
    _Float16* wpk1 = (_Float16*)sm; sm += 32 * 32  * 2;
    _Float16* wpk2 = (_Float16*)sm; sm += 32 * 288 * 2;
    _Float16* wpk3 = (_Float16*)sm; sm += 64 * 288 * 2;
    _Float16* wpk4 = (_Float16*)sm; sm += 64 * 576 * 2;
    float* pooled  = (float*)sm;    sm += 128 * 64  * 4;
    float* z       = (float*)sm;    sm += 128 * 512 * 4;
    float* h1e     = (float*)sm;    sm += 128 * 512 * 4;
    int*   top1    = (int*)sm;

    // Pack conv weights to f16 [COUT][KPAD] (K zero-padded to mult of 32).
    pack_weights<<<(32 * 32  + 255) / 256, 256, 0, stream>>>(c1w, wpk1, 32, 27,  32);
    pack_weights<<<(32 * 288 + 255) / 256, 256, 0, stream>>>(c2w, wpk2, 32, 288, 288);
    pack_weights<<<(64 * 288 + 255) / 256, 256, 0, stream>>>(c3w, wpk3, 64, 288, 288);
    pack_weights<<<(64 * 576 + 255) / 256, 256, 0, stream>>>(c4w, wpk4, 64, 576, 576);

    // x -> f16
    f32_to_f16<<<6291456 / 256, 256, 0, stream>>>(x, bufC, 6291456);

    // Encoder (WMMA convs + pools).
    conv3x3_wmma< 3, 32, 128, 128><<<128 * 128 * 2, 256, 0, stream>>>(bufC, wpk1, c1b, bufA);
    conv3x3_wmma<32, 32, 128, 128><<<128 * 128 * 2, 256, 0, stream>>>(bufA, wpk2, c2b, bufB);
    avgpool2_f16<<<(16777216 + 255) / 256, 256, 0, stream>>>(bufB, bufC, 32, 128, 128, 16777216);
    conv3x3_wmma<32, 64,  64,  64><<<128 * 64 * 2, 256, 0, stream>>>(bufC, wpk3, c3b, bufA);
    avgpool2_f16<<<(8388608 + 255) / 256, 256, 0, stream>>>(bufA, bufB, 64, 64, 64, 8388608);
    conv3x3_wmma<64, 64,  32,  32><<<128 * 32 * 1, 256, 0, stream>>>(bufB, wpk4, c4b, bufA);
    gmean_f16<<<8192, 256, 0, stream>>>(bufA, pooled);

    // Projection + router + experts.
    proj_kernel<<<256, 256, 0, stream>>>(pooled, pw, pb, z);
    router_kernel<<<1, 128, 0, stream>>>(z, rw, rb, top1, out + 128000);
    expert_fc1<<<128, 256, 0, stream>>>(z, w1, b1, top1, h1e);
    expert_fc2<<<128, 256, 0, stream>>>(h1e, w2, b2, top1, out);
}
